// GNN_695784702024
// MI455X (gfx1250) — compile-verified, matrix operand order checked by reading, not run on hardware
//
#include <hip/hip_runtime.h>

// ---------------- constants ----------------
#define NN     100000      // nodes
#define SS     2           // layout samples
#define GG     8           // graphs
#define EE     400000      // edges
#define MID    256
#define HALF   128
#define NL     4
#define MM     (SS * NN)   // 200000 rows
#define KPRE   288         // 265 padded to multiple of 32
#define APAD   296         // LDS row stride (shorts): 148 words % 64 = 20 -> conflict-free
#define INCH   265

typedef __attribute__((ext_vector_type(16))) __bf16 v16bf;
typedef __attribute__((ext_vector_type(8)))  float  v8f;
typedef __attribute__((ext_vector_type(4)))  int    i4;

struct Frag { i4 lo; i4 hi; };

static __device__ __forceinline__ v16bf make_frag(i4 lo, i4 hi) {
  Frag f; f.lo = lo; f.hi = hi;
  return __builtin_bit_cast(v16bf, f);
}

static __device__ __forceinline__ unsigned short f2bf(float f) {
  unsigned u = __builtin_bit_cast(unsigned, f);
  u += 0x7FFFu + ((u >> 16) & 1u);
  return (unsigned short)(u >> 16);
}
static __device__ __forceinline__ float bf2f(unsigned short s) {
  unsigned u = ((unsigned)s) << 16;
  return __builtin_bit_cast(float, u);
}
static __device__ __forceinline__ v8f vzero8() {
  v8f z;
#pragma unroll
  for (int i = 0; i < 8; i++) z[i] = 0.0f;
  return z;
}

// ---------------- weight prep ----------------
__global__ __launch_bounds__(256) void prep_pre(const float* __restrict__ preW,
                                                unsigned short* __restrict__ preWT) {
  int t = blockIdx.x * 256 + threadIdx.x;
  if (t >= MID * KPRE) return;
  int o = t / KPRE, k = t - o * KPRE;
  float v = (k < INCH) ? preW[k * MID + o] : 0.0f;
  preWT[t] = f2bf(v);
}

// WT[((l*2+d)*HALF + n)*512 + k]: k<256 -> Wl[l][k][n], else Wr[l][k-256][n]
__global__ __launch_bounds__(256) void prep_layers(const float* __restrict__ convWl,
                                                   const float* __restrict__ convWr,
                                                   const float* __restrict__ revWl,
                                                   const float* __restrict__ revWr,
                                                   unsigned short* __restrict__ WT) {
  int t = blockIdx.x * 256 + threadIdx.x;
  if (t >= NL * 2 * HALF * 512) return;
  int k = t & 511;
  int rest = t >> 9;
  int n = rest & (HALF - 1);
  rest >>= 7;
  int d = rest & 1;
  int l = rest >> 1;
  const float* Wl = d ? revWl : convWl;
  const float* Wr = d ? revWr : convWr;
  float v = (k < MID) ? Wl[(l * MID + k) * HALF + n]
                      : Wr[(l * MID + (k - MID)) * HALF + n];
  WT[t] = f2bf(v);
}

// ---------------- CSR build ----------------
__global__ __launch_bounds__(256) void cnt_zero(int* __restrict__ cF, int* __restrict__ cR) {
  int n = blockIdx.x * 256 + threadIdx.x;
  if (n < NN) { cF[n] = 0; cR[n] = 0; }
}
__global__ __launch_bounds__(256) void cnt_count(const int* __restrict__ src,
                                                 const int* __restrict__ tgt,
                                                 int* __restrict__ cF, int* __restrict__ cR) {
  int e = blockIdx.x * 256 + threadIdx.x;
  if (e >= EE) return;
  atomicAdd(&cF[tgt[e]], 1);   // fwd: rows keyed by tgt
  atomicAdd(&cR[src[e]], 1);   // rev: rows keyed by src
}
// single-block sequential-chunk exclusive scan (n up to a few 100k)
__global__ __launch_bounds__(256) void scan_k(const int* __restrict__ counts,
                                              int* __restrict__ rowp, int n) {
  __shared__ int sbuf[256];
  __shared__ int scarry;
  if (threadIdx.x == 0) scarry = 0;
  __syncthreads();
  for (int base = 0; base < n; base += 256) {
    int i = base + (int)threadIdx.x;
    int v = (i < n) ? counts[i] : 0;
    sbuf[threadIdx.x] = v;
    __syncthreads();
#pragma unroll
    for (int ofs = 1; ofs < 256; ofs <<= 1) {
      int tv = (threadIdx.x >= (unsigned)ofs) ? sbuf[threadIdx.x - ofs] : 0;
      __syncthreads();
      sbuf[threadIdx.x] += tv;
      __syncthreads();
    }
    if (i < n) rowp[i] = scarry + sbuf[threadIdx.x] - v;
    __syncthreads();
    if (threadIdx.x == 255) scarry += sbuf[255];
    __syncthreads();
  }
  if (threadIdx.x == 0) rowp[n] = scarry;
}
__global__ __launch_bounds__(256) void cursor_copy(const int* __restrict__ rowp,
                                                   int* __restrict__ cursor) {
  int n = blockIdx.x * 256 + threadIdx.x;
  if (n < NN) cursor[n] = rowp[n];
}
__global__ __launch_bounds__(256) void csr_fill(const int* __restrict__ key,
                                                const int* __restrict__ val,
                                                int* __restrict__ cursor,
                                                int* __restrict__ csr) {
  int e = blockIdx.x * 256 + threadIdx.x;
  if (e >= EE) return;
  int pos = atomicAdd(&cursor[key[e]], 1);
  csr[pos] = val[e];
}

// ---------------- fused feature assembly + pre GEMM ----------------
// block: 64 rows x 256 cols; 8 waves = 2 row-groups(32 rows) x 4 col-groups(64 cols)
// each wave: 2 A fragments, B fragment reused for both -> 2x arithmetic intensity
__global__ __launch_bounds__(256) void pre_gemm(const float* __restrict__ x_feat,
                                                const float* __restrict__ dim_feat,
                                                const float* __restrict__ layout_feat,
                                                const float* __restrict__ tile_feat,
                                                const float* __restrict__ opcode_embed,
                                                const int* __restrict__ node_opcode,
                                                const int* __restrict__ batch,
                                                const unsigned short* __restrict__ preWT,
                                                const float* __restrict__ preb,
                                                unsigned short* __restrict__ xout) {
  __shared__ alignas(16) unsigned short sA[64 * APAD];
  const int m0 = blockIdx.x * 64;
  const int tid = threadIdx.x;

  for (int idx = tid; idx < 64 * KPRE; idx += 256) {
    int r = idx / KPRE, k = idx - r * KPRE;
    int m = m0 + r;
    int s = (m >= NN) ? 1 : 0;
    int n = m - s * NN;
    float v = 0.0f;
    if (k < 53)        v = x_feat[n * 53 + k];
    else if (k < 85)   v = opcode_embed[node_opcode[n] * 32 + (k - 53)];
    else if (k < 223)  v = dim_feat[n * 138 + (k - 85)];
    else if (k < 247)  v = layout_feat[(n * SS + s) * 24 + (k - 223)];
    else if (k < INCH) v = tile_feat[(batch[n] * SS + s) * 18 + (k - 247)];
    sA[r * APAD + k] = f2bf(v);
  }
  __syncthreads();

  const int wid = tid >> 5, lane = tid & 31;
  const int rg = wid >> 2;   // row group (32 rows)
  const int cg = wid & 3;    // col group (64 cols)
  const int lr = lane & 15, hi = lane >> 4;

  v8f acc[2][4];
#pragma unroll
  for (int rt = 0; rt < 2; rt++)
#pragma unroll
    for (int ct = 0; ct < 4; ct++) acc[rt][ct] = vzero8();

  const unsigned short* sArow0 = sA + (rg * 32 + lr) * APAD;
  const unsigned short* sArow1 = sArow0 + 16 * APAD;
#pragma unroll
  for (int kk = 0; kk < KPRE; kk += 32) {
    v16bf a0 = make_frag(*(const i4*)(sArow0 + kk + hi * 8),
                         *(const i4*)(sArow0 + kk + 16 + hi * 8));
    v16bf a1 = make_frag(*(const i4*)(sArow1 + kk + hi * 8),
                         *(const i4*)(sArow1 + kk + 16 + hi * 8));
#pragma unroll
    for (int ct = 0; ct < 4; ct++) {
      const unsigned short* bp = preWT + (cg * 64 + ct * 16 + lr) * KPRE + kk + hi * 8;
      v16bf b = make_frag(*(const i4*)bp, *(const i4*)(bp + 16));
      acc[0][ct] = __builtin_amdgcn_wmma_f32_16x16x32_bf16(
          false, a0, false, b, (short)0, acc[0][ct], false, false);
      acc[1][ct] = __builtin_amdgcn_wmma_f32_16x16x32_bf16(
          false, a1, false, b, (short)0, acc[1][ct], false, false);
    }
  }

#pragma unroll
  for (int rt = 0; rt < 2; rt++)
#pragma unroll
    for (int ct = 0; ct < 4; ct++) {
      int col = cg * 64 + ct * 16 + lr;
      float bias = preb[col];
#pragma unroll
      for (int r = 0; r < 8; r++) {
        int row = m0 + rg * 32 + rt * 16 + hi * 8 + r;
        float v = fmaxf(acc[rt][ct][r] + bias, 0.0f);
        xout[row * MID + col] = f2bf(v);
      }
    }
}

// ---------------- CSR gather-reduce aggregation (no atomics) ----------------
// wave = one node, 32 lanes own 8-channel slices; each neighbor row is a
// coalesced 512B wave read. Mean-scaled, bf16 output, both samples per pass.
__global__ __launch_bounds__(256) void aggregate_k(const unsigned short* __restrict__ x,
                                                   const int* __restrict__ rowp,
                                                   const int* __restrict__ cols,
                                                   unsigned short* __restrict__ out) {
  int t = blockIdx.x * 256 + threadIdx.x;
  if (t >= NN * 32) return;
  int n = t >> 5;
  int c = (t & 31) * 8;
  int beg = rowp[n], end = rowp[n + 1];
  float inv = 1.0f / fmaxf((float)(end - beg), 1.0f);
  float acc[SS][8];
#pragma unroll
  for (int s = 0; s < SS; s++)
#pragma unroll
    for (int j = 0; j < 8; j++) acc[s][j] = 0.0f;

  for (int e = beg; e < end; e++) {
    int nb = cols[e];
#pragma unroll
    for (int s = 0; s < SS; s++) {
      const i4 v = *(const i4*)(x + (size_t)(s * NN + nb) * MID + c);
#pragma unroll
      for (int j = 0; j < 4; j++) {
        unsigned w = (unsigned)v[j];
        acc[s][2 * j + 0] += bf2f((unsigned short)(w & 0xffffu));
        acc[s][2 * j + 1] += bf2f((unsigned short)(w >> 16));
      }
    }
  }
#pragma unroll
  for (int s = 0; s < SS; s++) {
    i4 o;
#pragma unroll
    for (int j = 0; j < 4; j++) {
      unsigned lo = f2bf(acc[s][2 * j + 0] * inv);
      unsigned hh = f2bf(acc[s][2 * j + 1] * inv);
      o[j] = (int)(lo | (hh << 16));
    }
    *(i4*)(out + (size_t)(s * NN + n) * MID + c) = o;
  }
}

// ---------------- layer GEMM ----------------
// block: 64 rows; 8 waves = half(2) x row-group(2: 32 rows) x col-group(2: 64 cols)
// each wave: 2 A fragments share every B fragment (2 WMMAs per B fetch)
__global__ __launch_bounds__(256) void layer_gemm(const unsigned short* __restrict__ xin,
                                                  const unsigned short* __restrict__ aggF,
                                                  const unsigned short* __restrict__ aggR,
                                                  const unsigned short* __restrict__ WT,
                                                  int l,
                                                  const float* __restrict__ convb_l,
                                                  const float* __restrict__ revb_l,
                                                  unsigned short* __restrict__ xout) {
  const int m0 = blockIdx.x * 64;
  const int tid = threadIdx.x, wid = tid >> 5, lane = tid & 31;
  const int half = wid >> 2;
  const int rg = (wid >> 1) & 1;
  const int cgp = wid & 1;
  const int lr = lane & 15, hi = lane >> 4;

  const unsigned short* aggp = half ? aggR : aggF;
  const unsigned short* wbase = WT + (size_t)((l * 2 + half) * HALF) * 512;
  const int rbase = m0 + rg * 32;

  v8f acc[2][4];
#pragma unroll
  for (int rt = 0; rt < 2; rt++)
#pragma unroll
    for (int ct = 0; ct < 4; ct++) acc[rt][ct] = vzero8();

#pragma unroll
  for (int kk = 0; kk < 512; kk += 32) {
    const unsigned short* asrc = (kk < 256) ? aggp : xin;
    int kke = kk & 255;
    const unsigned short* ap0 = asrc + (size_t)(rbase + lr) * MID + kke + hi * 8;
    const unsigned short* ap1 = ap0 + 16 * MID;
    v16bf a0 = make_frag(*(const i4*)ap0, *(const i4*)(ap0 + 16));
    v16bf a1 = make_frag(*(const i4*)ap1, *(const i4*)(ap1 + 16));
#pragma unroll
    for (int ct = 0; ct < 4; ct++) {
      const unsigned short* bp =
          wbase + (size_t)(cgp * 64 + ct * 16 + lr) * 512 + kk + hi * 8;
      v16bf b = make_frag(*(const i4*)bp, *(const i4*)(bp + 16));
      acc[0][ct] = __builtin_amdgcn_wmma_f32_16x16x32_bf16(
          false, a0, false, b, (short)0, acc[0][ct], false, false);
      acc[1][ct] = __builtin_amdgcn_wmma_f32_16x16x32_bf16(
          false, a1, false, b, (short)0, acc[1][ct], false, false);
    }
  }

  const float* biasp = half ? revb_l : convb_l;
#pragma unroll
  for (int rt = 0; rt < 2; rt++)
#pragma unroll
    for (int ct = 0; ct < 4; ct++) {
      int colh = cgp * 64 + ct * 16 + lr;
      float bias = biasp[colh];
      int col = half * HALF + colh;
#pragma unroll
      for (int r = 0; r < 8; r++) {
        int row = m0 + rg * 32 + rt * 16 + hi * 8 + r;
        float v = fmaxf(acc[rt][ct][r] + bias, 0.0f);
        xout[(size_t)row * MID + col] = f2bf(v);
      }
    }
}

// ---------------- pooling + head ----------------
__global__ __launch_bounds__(256) void pool_zero(float* __restrict__ pooled) {
  int t = blockIdx.x * 256 + threadIdx.x;
  if (t < SS * GG * 512) pooled[t] = 0.0f;
}

// grid.x = row chunks, grid.y = 64 channel-octets; LDS bins then sparse flush
__global__ __launch_bounds__(256) void pool_k(const unsigned short* __restrict__ x3,
                                              const unsigned short* __restrict__ x4,
                                              const int* __restrict__ batch,
                                              float* __restrict__ pooled) {
  __shared__ float bins[SS * GG * 8];   // 128 floats
  const int cc = blockIdx.y;            // 0..63
  const int m = blockIdx.x * 256 + threadIdx.x;
  if (threadIdx.x < 128) bins[threadIdx.x] = 0.0f;
  __syncthreads();
  if (m < MM) {
    int c = cc * 8;
    int s = (m >= NN) ? 1 : 0;
    int n = m - s * NN;
    int g = batch[n];
    const unsigned short* srcp =
        (c < MID) ? (x3 + (size_t)m * MID + c) : (x4 + (size_t)m * MID + (c - MID));
    const i4 v = *(const i4*)srcp;
    float* bp = bins + (s * GG + g) * 8;
#pragma unroll
    for (int j = 0; j < 4; j++) {
      unsigned w = (unsigned)v[j];
      atomicAdd(bp + 2 * j + 0, bf2f((unsigned short)(w & 0xffffu)));
      atomicAdd(bp + 2 * j + 1, bf2f((unsigned short)(w >> 16)));
    }
  }
  __syncthreads();
  if (threadIdx.x < 128) {
    float v = bins[threadIdx.x];
    if (v != 0.0f) {
      int sg = threadIdx.x >> 3, j = threadIdx.x & 7;
      atomicAdd(&pooled[(size_t)sg * 512 + cc * 8 + j], v);
    }
  }
}

__global__ __launch_bounds__(32) void head_k(const float* __restrict__ pooled,
                                             const float* __restrict__ headW,
                                             const float* __restrict__ headb,
                                             float* __restrict__ out) {
  int t = threadIdx.x;
  if (t >= GG * SS) return;
  int g = t >> 1, s = t & 1;
  const float* pp = pooled + (size_t)(s * GG + g) * 512;
  float sum = headb[0];
  for (int c = 0; c < 512; c++) sum += pp[c] * headW[c];
  out[g * SS + s] = sum;
}

// ---------------- launch ----------------
extern "C" void kernel_launch(void* const* d_in, const int* in_sizes, int n_in,
                              void* d_out, int out_size, void* d_ws, size_t ws_size,
                              hipStream_t stream) {
  const float* x_feat       = (const float*)d_in[0];
  const float* dim_feat     = (const float*)d_in[1];
  const float* layout_feat  = (const float*)d_in[2];
  const float* tile_feat    = (const float*)d_in[3];
  const float* opcode_embed = (const float*)d_in[4];
  const float* preW         = (const float*)d_in[5];
  const float* preb         = (const float*)d_in[6];
  const float* convWl       = (const float*)d_in[7];
  const float* convWr       = (const float*)d_in[8];
  const float* convb        = (const float*)d_in[9];
  const float* revWl        = (const float*)d_in[10];
  const float* revWr        = (const float*)d_in[11];
  const float* revb         = (const float*)d_in[12];
  const float* headW        = (const float*)d_in[13];
  const float* headb        = (const float*)d_in[14];
  const int* node_opcode    = (const int*)d_in[15];
  const int* batch          = (const int*)d_in[16];
  const int* edge_index     = (const int*)d_in[17];
  const int* srcI = edge_index;
  const int* tgtI = edge_index + EE;

  char* p = (char*)d_ws;
  auto carve = [&](size_t bytes) -> char* {
    char* q = p;
    p += (bytes + 255) & ~(size_t)255;
    return q;
  };
  unsigned short* xA    = (unsigned short*)carve((size_t)MM * MID * 2);
  unsigned short* xB    = (unsigned short*)carve((size_t)MM * MID * 2);
  unsigned short* aggFb = (unsigned short*)carve((size_t)MM * MID * 2);
  unsigned short* aggRb = (unsigned short*)carve((size_t)MM * MID * 2);
  unsigned short* preWT = (unsigned short*)carve((size_t)MID * KPRE * 2);
  unsigned short* WT    = (unsigned short*)carve((size_t)NL * 2 * HALF * 512 * 2);
  int* cntF   = (int*)carve((size_t)NN * 4);         // reused as cursor
  int* cntR   = (int*)carve((size_t)NN * 4);
  int* rowF   = (int*)carve((size_t)(NN + 1) * 4);
  int* rowR   = (int*)carve((size_t)(NN + 1) * 4);
  int* csrF   = (int*)carve((size_t)EE * 4);
  int* csrR   = (int*)carve((size_t)EE * 4);
  float* pooled = (float*)carve((size_t)SS * GG * 512 * 4);

  // weight prep
  prep_pre<<<(MID * KPRE + 255) / 256, 256, 0, stream>>>(preW, preWT);
  prep_layers<<<(NL * 2 * HALF * 512 + 255) / 256, 256, 0, stream>>>(
      convWl, convWr, revWl, revWr, WT);

  // CSR build (fwd keyed by tgt holding src; rev keyed by src holding tgt)
  cnt_zero<<<(NN + 255) / 256, 256, 0, stream>>>(cntF, cntR);
  cnt_count<<<(EE + 255) / 256, 256, 0, stream>>>(srcI, tgtI, cntF, cntR);
  scan_k<<<1, 256, 0, stream>>>(cntF, rowF, NN);
  scan_k<<<1, 256, 0, stream>>>(cntR, rowR, NN);
  cursor_copy<<<(NN + 255) / 256, 256, 0, stream>>>(rowF, cntF);
  cursor_copy<<<(NN + 255) / 256, 256, 0, stream>>>(rowR, cntR);
  csr_fill<<<(EE + 255) / 256, 256, 0, stream>>>(tgtI, srcI, cntF, csrF);
  csr_fill<<<(EE + 255) / 256, 256, 0, stream>>>(srcI, tgtI, cntR, csrR);

  // feature assembly + pre-linear (WMMA)
  pre_gemm<<<MM / 64, 256, 0, stream>>>(x_feat, dim_feat, layout_feat, tile_feat,
                                        opcode_embed, node_opcode, batch,
                                        preWT, preb, xA);

  unsigned short* cur = xA;
  unsigned short* nxt = xB;
  for (int l = 0; l < NL; l++) {
    aggregate_k<<<(NN * 32 + 255) / 256, 256, 0, stream>>>(cur, rowF, csrF, aggFb);
    aggregate_k<<<(NN * 32 + 255) / 256, 256, 0, stream>>>(cur, rowR, csrR, aggRb);
    layer_gemm<<<MM / 64, 256, 0, stream>>>(cur, aggFb, aggRb, WT, l,
                                            convb + l * HALF, revb + l * HALF, nxt);
    unsigned short* t = cur; cur = nxt; nxt = t;
  }
  // after 4 layers: cur = x4, nxt = x3

  pool_zero<<<(SS * GG * 512 + 255) / 256, 256, 0, stream>>>(pooled);
  dim3 pg((MM + 255) / 256, 64);
  pool_k<<<pg, 256, 0, stream>>>(nxt, cur, batch, pooled);
  head_k<<<1, 32, 0, stream>>>(pooled, headW, headb, (float*)d_out);
}
